// EdgeEvidenceHead_26603027432198
// MI455X (gfx1250) — matrix operand order repaired and checked by reading, not compiled, not run on hardware
//
#include <hip/hip_runtime.h>
#include <hip/hip_bf16.h>
#include <math.h>

typedef _Float16 v16h __attribute__((ext_vector_type(16)));
typedef _Float16 v8h  __attribute__((ext_vector_type(8)));
typedef _Float16 v4h  __attribute__((ext_vector_type(4)));
typedef float    v8f  __attribute__((ext_vector_type(8)));

#define HIDDEN   128
#define EHID     256
#define FEATN    519
#define KPAD     544      // 17 * 32
#define KSTEPS   17
#define NTILES   16       // 256 / 16
#define FSTRIDE  552      // padded LDS row stride (halfs); 1104B keeps A-frag b128 loads conflict-free
#define MT       64       // edges per workgroup
#define THREADS  256

#define LDS_FEAT_BYTES (MT * FSTRIDE * 2)   // 70656 (h[64][256] f32 = 65536 aliases into this)
#define LDS_RED_BYTES  (MT * 4 * 3 * 4)     // 3072
#define LDS_W2_BYTES   (EHID * 2 * 4)       // 2048
#define SMEM_BYTES (LDS_FEAT_BYTES + LDS_RED_BYTES + LDS_W2_BYTES)

union AB16 { v16h v; v8h h[2]; };

__device__ __forceinline__ float softplus_f(float x) {
    return (x > 20.f) ? x : log1pf(expf(x));
}

// Pack W1 [519,256] fp32 -> f16, K zero-padded to 544, in per-lane WMMA
// B-fragment order: fragment (ktile s, ntile j): lane l holds column n=j*16+(l&15),
// K = s*32 + (l<16 ? 0 : 16) + i, i=0..15 (contiguous K order, matching the A fragments).
__global__ __launch_bounds__(256) void pack_w1_kernel(const float* __restrict__ W1,
                                                      _Float16* __restrict__ Wpk) {
    int t = blockIdx.x * blockDim.x + threadIdx.x;      // 0 .. 17*16*32-1
    if (t >= KSTEPS * NTILES * 32) return;
    int lane = t & 31;
    int j    = (t >> 5) & 15;
    int s    = t >> 9;
    int n    = j * 16 + (lane & 15);
    int kb   = s * 32 + ((lane < 16) ? 0 : 16);
    v16h o;
#pragma unroll
    for (int i = 0; i < 16; ++i) {
        int k = kb + i;
        float w = (k < FEATN) ? W1[k * EHID + n] : 0.0f;
        o[i] = (_Float16)w;
    }
    *(v16h*)(Wpk + (size_t)t * 16) = o;
}

__global__ __launch_bounds__(THREADS) void EdgeEvidenceHead_kernel(
    const float* __restrict__ z, const long long* __restrict__ eidx,
    const float* __restrict__ extra, const float* __restrict__ b1,
    const float* __restrict__ W2, const float* __restrict__ b2,
    const _Float16* __restrict__ Wpk, float* __restrict__ out, int E)
{
    extern __shared__ char smem[];
    _Float16* feat = (_Float16*)smem;                           // [64][552] f16
    float*    hlds = (float*)smem;                              // alias: [64][256] f32 (after GEMM1)
    float*    red  = (float*)(smem + LDS_FEAT_BYTES);           // 768 floats
    float*    w2l  = (float*)(smem + LDS_FEAT_BYTES + LDS_RED_BYTES); // 512 floats

    const int tid  = threadIdx.x;
    const int lane = tid & 31;
    const int wave = tid >> 5;
    const int e0   = blockIdx.x * MT;

    // stage W2 into LDS (broadcast-hot in stage 2)
    w2l[tid]       = W2[tid];
    w2l[tid + 256] = W2[tid + 256];

    // ---------------- Feature build: 4 threads per edge, 32 dims each -------------
    {
        int m = tid >> 2;
        int p = tid & 3;
        int d0 = p * 32;
        int e = e0 + m;
        _Float16* frow = feat + m * FSTRIDE;
        float dotp = 0.f, nu2 = 0.f, nv2 = 0.f;
        if (e < E) {
            long long u = eidx[e];
            long long v = eidx[(long long)E + e];
            const float4* zu4 = (const float4*)(z + (size_t)u * HIDDEN + d0);
            const float4* zv4 = (const float4*)(z + (size_t)v * HIDDEN + d0);
#pragma unroll
            for (int i = 0; i < 8; ++i) {
                float4 a = zu4[i];
                float4 b = zv4[i];
                float ax[4] = {a.x, a.y, a.z, a.w};
                float bx[4] = {b.x, b.y, b.z, b.w};
                v4h ha, hb, hd, hp;
#pragma unroll
                for (int c = 0; c < 4; ++c) {
                    float av = ax[c], bv = bx[c];
                    float pr = av * bv;
                    ha[c] = (_Float16)av;
                    hb[c] = (_Float16)bv;
                    hd[c] = (_Float16)fabsf(av - bv);
                    hp[c] = (_Float16)pr;
                    dotp += pr; nu2 += av * av; nv2 += bv * bv;
                }
                int o = d0 + i * 4;
                *(v4h*)(frow + o)              = ha;
                *(v4h*)(frow + HIDDEN + o)     = hb;
                *(v4h*)(frow + 2 * HIDDEN + o) = hd;
                *(v4h*)(frow + 3 * HIDDEN + o) = hp;
            }
        } else {
            v4h hz = {(_Float16)0.f, (_Float16)0.f, (_Float16)0.f, (_Float16)0.f};
#pragma unroll
            for (int i = 0; i < 8; ++i) {
                int o = d0 + i * 4;
                *(v4h*)(frow + o)              = hz;
                *(v4h*)(frow + HIDDEN + o)     = hz;
                *(v4h*)(frow + 2 * HIDDEN + o) = hz;
                *(v4h*)(frow + 3 * HIDDEN + o) = hz;
            }
        }
        red[m * 4 + p]       = dotp;
        red[256 + m * 4 + p] = nu2;
        red[512 + m * 4 + p] = nv2;
    }
    __syncthreads();
    {
        int m = tid >> 2;
        int p = tid & 3;
        if (p == 0) {
            int e = e0 + m;
            _Float16* frow = feat + m * FSTRIDE;
            float dotv = red[m*4] + red[m*4+1] + red[m*4+2] + red[m*4+3];
            float nu2  = red[256+m*4] + red[256+m*4+1] + red[256+m*4+2] + red[256+m*4+3];
            float nv2  = red[512+m*4] + red[512+m*4+1] + red[512+m*4+2] + red[512+m*4+3];
            float nu = fmaxf(sqrtf(nu2), 1e-12f);
            float nv = fmaxf(sqrtf(nv2), 1e-12f);
            float cosv = dotv / (nu * nv);
            frow[512] = (_Float16)dotv;
            frow[513] = (_Float16)cosv;
            if (e < E) {
#pragma unroll
                for (int j = 0; j < 5; ++j)
                    frow[514 + j] = (_Float16)extra[(size_t)e * 5 + j];
            } else {
#pragma unroll
                for (int j = 0; j < 5; ++j) frow[514 + j] = (_Float16)0.f;
            }
            for (int k = FEATN; k < KPAD; ++k) frow[k] = (_Float16)0.f;
        }
    }
    __syncthreads();

    // ---------------- GEMM1: feat[64x544] @ W1[544x256], wave owns 32 cols --------
    // Fixed 17-step K loop, fully unrolled: no backedge phis, so accumulators chain
    // WMMA->WMMA directly (no v_mov_b64 rotation), and next-step ds/global loads
    // schedule under the current step's 8 WMMAs.
    v8f vzero = {0.f, 0.f, 0.f, 0.f, 0.f, 0.f, 0.f, 0.f};
    v8f acc[4][2];
#pragma unroll
    for (int mt = 0; mt < 4; ++mt)
#pragma unroll
        for (int nt = 0; nt < 2; ++nt) acc[mt][nt] = vzero;

    const int rsel = lane & 15;
    const int kb   = (lane < 16) ? 0 : 16;  // A-frag contiguous K order (matches packed B)
    const int j0   = wave * 2;
    const _Float16* fbase = feat + rsel * FSTRIDE + kb;
    const _Float16* wbase = Wpk + (size_t)((j0 * 32 + lane) * 16);

#pragma unroll
    for (int s = 0; s < KSTEPS; ++s) {
        const int k0 = s * 32;
        AB16 a[4];
#pragma unroll
        for (int mt = 0; mt < 4; ++mt) {
            const _Float16* fp = fbase + mt * 16 * FSTRIDE + k0;
            a[mt].h[0] = *(const v8h*)fp;          // K = kb+0..7
            a[mt].h[1] = *(const v8h*)(fp + 8);    // K = kb+8..15
        }
        v16h bfr[2];
#pragma unroll
        for (int nt = 0; nt < 2; ++nt)
            bfr[nt] = *(const v16h*)(wbase + (size_t)(s * NTILES + nt) * 32 * 16);
#pragma unroll
        for (int mt = 0; mt < 4; ++mt)
#pragma unroll
            for (int nt = 0; nt < 2; ++nt)
                acc[mt][nt] = __builtin_amdgcn_wmma_f32_16x16x32_f16(
                    false, a[mt].v, false, bfr[nt], (short)0, acc[mt][nt], false, false);
    }
    __syncthreads();   // everyone done reading feat before aliasing as hlds

    // ---------------- bias + ReLU, write h to LDS (C/D layout, ISA 7.12.2) --------
    {
        int madd = (lane < 16) ? 0 : 8;
#pragma unroll
        for (int nt = 0; nt < 2; ++nt) {
            int n = (j0 + nt) * 16 + rsel;
            float bias = b1[n];
#pragma unroll
            for (int mt = 0; mt < 4; ++mt) {
#pragma unroll
                for (int i = 0; i < 8; ++i) {
                    int row = mt * 16 + madd + i;
                    hlds[row * EHID + n] = fmaxf(acc[mt][nt][i] + bias, 0.f);
                }
            }
        }
    }
    __syncthreads();

    // ---------------- GEMM2 (N=2) + softplus in VALU ------------------------------
    {
        int m = tid & 63;
        int p = tid >> 6;
        float s0 = 0.f, s1 = 0.f;
#pragma unroll 4
        for (int i = 0; i < 64; ++i) {
            int k = p * 64 + ((i + m) & 63);   // rotation: conflict-free hlds reads
            float hv = hlds[m * EHID + k];
            s0 += hv * w2l[k * 2];
            s1 += hv * w2l[k * 2 + 1];
        }
        red[m * 4 + p]       = s0;
        red[256 + m * 4 + p] = s1;
    }
    __syncthreads();
    if (tid < MT) {
        int m = tid;
        int e = e0 + m;
        if (e < E) {
            float s0 = red[m*4] + red[m*4+1] + red[m*4+2] + red[m*4+3] + b2[0];
            float s1 = red[256+m*4] + red[256+m*4+1] + red[256+m*4+2] + red[256+m*4+3] + b2[1];
            out[(size_t)e * 2]     = softplus_f(s0);
            out[(size_t)e * 2 + 1] = softplus_f(s1);
        }
    }
}

extern "C" void kernel_launch(void* const* d_in, const int* in_sizes, int n_in,
                              void* d_out, int out_size, void* d_ws, size_t ws_size,
                              hipStream_t stream) {
    const float*     z     = (const float*)d_in[0];
    const long long* eidx  = (const long long*)d_in[1];
    const float*     extra = (const float*)d_in[2];
    const float*     W1    = (const float*)d_in[3];
    const float*     b1    = (const float*)d_in[4];
    const float*     W2    = (const float*)d_in[5];
    const float*     b2    = (const float*)d_in[6];
    float*           out   = (float*)d_out;
    const int E = in_sizes[2] / 5;           // extra_feat is [E,5]
    _Float16* Wpk = (_Float16*)d_ws;         // 544*256 f16 = 278528 B

    int packN = KSTEPS * NTILES * 32;        // 8704 threads, one v16h each
    pack_w1_kernel<<<(packN + 255) / 256, 256, 0, stream>>>(W1, Wpk);

    int nb = (E + MT - 1) / MT;
    EdgeEvidenceHead_kernel<<<nb, THREADS, SMEM_BYTES, stream>>>(
        z, eidx, extra, b1, W2, b2, Wpk, out, E);
}
